// GeneralGCN_52450140619485
// MI455X (gfx1250) — matrix operand order repaired and checked by reading, not compiled, not run on hardware
//
#include <hip/hip_runtime.h>
#include <hip/hip_bf16.h>

#define NN 50000
#define NE 800000
#define DIN 1024
#define DH  128
#define DO  9

typedef __attribute__((ext_vector_type(16))) _Float16 v16h;
typedef __attribute__((ext_vector_type(8)))  float    v8f;

// ---------------------------------------------------------------------------
// Pack a row-major f32 weight matrix W[K, ncols] into WMMA B-fragment order:
//   dst[((ks*NCT_total + ct)*32 + lane)*16 + i]  holds  B[k][n]  (f16)
//   with k = ks*32 + (lane/16)*16 + i,  n = ct*16 + (lane%16).
// Columns >= ncols (padding) are zero-filled.
// ---------------------------------------------------------------------------
__global__ void pack_b_kernel(const float* __restrict__ W, int K, int ncols,
                              _Float16* __restrict__ dst, int NCT_total, int ct0,
                              int nct_sub, int ksteps)
{
    int tid = blockIdx.x * blockDim.x + threadIdx.x;
    int total = ksteps * nct_sub * 512;
    if (tid >= total) return;
    int i    = tid & 15;
    int lane = (tid >> 4) & 31;
    int ctl  = (tid >> 9) % nct_sub;
    int ks   = tid / (512 * nct_sub);
    int half = lane >> 4;
    int k    = ks * 32 + half * 16 + i;          // B fragment: elem i <-> K = half*16 + i
    int col  = ctl * 16 + (lane & 15);
    float val = 0.f;
    if (k < K && col < ncols) val = W[(size_t)k * ncols + col];
    size_t di = ((size_t)(ks * NCT_total + (ct0 + ctl)) * 32 + lane) * 16 + i;
    dst[di] = (_Float16)val;
}

// ---------------------------------------------------------------------------
// WMMA GEMM: each wave computes a 16-row x (NCT*16)-col tile of
//   out = X[f32] @ Bpacked[f16]  (+ bias), accumulating in f32.
// Output columns [0, ncols0) -> out0 (ld0), the rest -> out1 (ld1).
// A fragment (16x32 f16): lane<16 row=lane holds K 0..15; lane>=16 K 16..31.
// ---------------------------------------------------------------------------
template<int NCT>
__global__ __launch_bounds__(256) void gemm_wmma_kernel(
    const float* __restrict__ X, int ldx, int ksteps, int rowTiles,
    const _Float16* __restrict__ Bp,
    const float* __restrict__ bias0, int nvalid0, int ncols0,
    const float* __restrict__ bias1, int nvalid1,
    float* __restrict__ out0, int ld0,
    float* __restrict__ out1, int ld1)
{
    int wave = blockIdx.x * (blockDim.x >> 5) + (threadIdx.x >> 5);
    if (wave >= rowTiles) return;                 // wave-uniform: EXEC stays all-1s for WMMA
    int lane = threadIdx.x & 31;
    int half = lane >> 4;
    int row  = wave * 16 + (lane & 15);
    const float* xrow = X + (size_t)row * ldx + half * 16;

    v8f acc[NCT] = {};

    for (int ks = 0; ks < ksteps; ++ks) {
        // ---- load + convert A fragment (f32 -> f16) ----
        v16h a;
        const float* p = xrow + ks * 32;
#pragma unroll
        for (int g = 0; g < 2; ++g) {             // K groups +0 / +8
            float4 f0 = *(const float4*)(p + g * 8);
            float4 f1 = *(const float4*)(p + g * 8 + 4);
            a[g*8+0] = (_Float16)f0.x; a[g*8+1] = (_Float16)f0.y;
            a[g*8+2] = (_Float16)f0.z; a[g*8+3] = (_Float16)f0.w;
            a[g*8+4] = (_Float16)f1.x; a[g*8+5] = (_Float16)f1.y;
            a[g*8+6] = (_Float16)f1.z; a[g*8+7] = (_Float16)f1.w;
        }
        const _Float16* bp = Bp + ((size_t)ks * NCT * 32 + lane) * 16;
#pragma unroll
        for (int c = 0; c < NCT; ++c) {
            v16h b = *(const v16h*)(bp + (size_t)c * 512);
            acc[c] = __builtin_amdgcn_wmma_f32_16x16x32_f16(
                false, a, false, b, (short)0, acc[c], false, false);
        }
    }

    // ---- bias + store: C elem v -> row (v + 8*half), col = lane%16 ----
#pragma unroll
    for (int c = 0; c < NCT; ++c) {
        int colg = c * 16 + (lane & 15);
        float bias; float* outp; int ld;
        if (colg < ncols0) {
            bias = (colg < nvalid0) ? bias0[colg] : 0.f;
            outp = out0 + colg; ld = ld0;
        } else {
            int c1 = colg - ncols0;
            bias = (c1 < nvalid1) ? bias1[c1] : 0.f;
            outp = out1 + c1; ld = ld1;
        }
#pragma unroll
        for (int v = 0; v < 8; ++v) {
            int r = wave * 16 + v + half * 8;
            outp[(size_t)r * ld] = acc[c][v] + bias;
        }
    }
}

// ---------------------------------------------------------------------------
// Attention pipeline kernels
// ---------------------------------------------------------------------------
__global__ void node_score_kernel(const float* __restrict__ m, int ldm, int F,
                                  const float* __restrict__ att,
                                  float* __restrict__ s, int n)
{
    int node = blockIdx.x * (blockDim.x >> 5) + (threadIdx.x >> 5);
    if (node >= n) return;
    int lane = threadIdx.x & 31;
    float acc = 0.f;
    for (int f = lane; f < F; f += 32) acc += m[(size_t)node * ldm + f] * att[f];
#pragma unroll
    for (int off = 16; off > 0; off >>= 1) acc += __shfl_xor(acc, off, 32);
    if (lane == 0) s[node] = acc > 0.f ? acc : 0.2f * acc;   // leaky_relu(dot)
}

__global__ void init_attn_kernel(float* __restrict__ amax, float* __restrict__ denom, int n)
{
    int i = blockIdx.x * blockDim.x + threadIdx.x;
    if (i >= n) return;
    amax[i]  = __int_as_float(0xFF800000);   // -inf
    denom[i] = 0.f;
}

__device__ __forceinline__ void atomicMaxF(float* addr, float val)
{
    if (val >= 0.f) atomicMax((int*)addr, __float_as_int(val));
    else            atomicMin((unsigned int*)addr, (unsigned int)__float_as_int(val));
}

__global__ void edge_max_kernel(const int* __restrict__ src, const int* __restrict__ dst,
                                const float* __restrict__ s, float* __restrict__ amax, int e)
{
    int i = blockIdx.x * blockDim.x + threadIdx.x;
    if (i >= e) return;
    atomicMaxF(&amax[dst[i]], s[src[i]]);
}

__global__ void edge_exp_kernel(const int* __restrict__ src, const int* __restrict__ dst,
                                const float* __restrict__ s, const float* __restrict__ amax,
                                float* __restrict__ denom, float* __restrict__ eexp, int e)
{
    int i = blockIdx.x * blockDim.x + threadIdx.x;
    if (i >= e) return;
    int d = dst[i];
    float ev = expf(s[src[i]] - amax[d]);
    eexp[i] = ev;
    atomicAdd(&denom[d], ev);
}

__global__ void alpha_kernel(const int* __restrict__ dst, const float* __restrict__ denom,
                             float* __restrict__ eexp, int e)
{
    int i = blockIdx.x * blockDim.x + threadIdx.x;
    if (i >= e) return;
    eexp[i] = eexp[i] / (denom[dst[i]] + 1e-16f);
}

__global__ void edge_scatter_kernel(const int* __restrict__ src, const int* __restrict__ dst,
                                    const float* __restrict__ m, int ldm,
                                    const float* __restrict__ alpha,
                                    float* __restrict__ acc, int ldacc, int F, int total)
{
    int i = blockIdx.x * blockDim.x + threadIdx.x;
    if (i >= total) return;
    int e = i / F, f = i - e * F;
    atomicAdd(&acc[(size_t)dst[e] * ldacc + f], m[(size_t)src[e] * ldm + f] * alpha[e]);
}

// h = relu( l2norm(acc) ), F = 128, wave per row
__global__ void combine_kernel(const float* __restrict__ acc, float* __restrict__ h, int n)
{
    int node = blockIdx.x * (blockDim.x >> 5) + (threadIdx.x >> 5);
    if (node >= n) return;
    int lane = threadIdx.x & 31;
    const float* row = acc + (size_t)node * DH;
    float v[4]; float ss = 0.f;
#pragma unroll
    for (int j = 0; j < 4; ++j) { v[j] = row[lane + 32 * j]; ss += v[j] * v[j]; }
#pragma unroll
    for (int off = 16; off > 0; off >>= 1) ss += __shfl_xor(ss, off, 32);
    float inv = 1.f / fmaxf(sqrtf(ss), 1e-12f);
    float* out = h + (size_t)node * DH;
#pragma unroll
    for (int j = 0; j < 4; ++j) out[lane + 32 * j] = fmaxf(v[j] * inv, 0.f);
}

// final: l2norm over 9, then log_softmax, thread per node
__global__ void final_kernel(const float* __restrict__ acc, float* __restrict__ out, int n)
{
    int node = blockIdx.x * blockDim.x + threadIdx.x;
    if (node >= n) return;
    const float* row = acc + (size_t)node * 16;
    float v[DO]; float ss = 0.f;
#pragma unroll
    for (int j = 0; j < DO; ++j) { v[j] = row[j]; ss += v[j] * v[j]; }
    float inv = 1.f / fmaxf(sqrtf(ss), 1e-12f);
    float mx = -1e30f;
#pragma unroll
    for (int j = 0; j < DO; ++j) { v[j] *= inv; mx = fmaxf(mx, v[j]); }
    float se = 0.f;
#pragma unroll
    for (int j = 0; j < DO; ++j) se += expf(v[j] - mx);
    float lse = mx + logf(se);
    float* o = out + (size_t)node * DO;
#pragma unroll
    for (int j = 0; j < DO; ++j) o[j] = v[j] - lse;
}

// ---------------------------------------------------------------------------
extern "C" void kernel_launch(void* const* d_in, const int* in_sizes, int n_in,
                              void* d_out, int out_size, void* d_ws, size_t ws_size,
                              hipStream_t stream)
{
    const float* x    = (const float*)d_in[0];
    const int*   ei   = (const int*)d_in[1];
    const int*   srcp = ei;            // edge_index[0] = source
    const int*   dstp = ei + NE;       // edge_index[1] = target
    const float* W1   = (const float*)d_in[2];
    const float* b1   = (const float*)d_in[3];
    const float* att1 = (const float*)d_in[4];
    const float* Ws1  = (const float*)d_in[5];
    const float* bs1  = (const float*)d_in[6];
    const float* W2   = (const float*)d_in[7];
    const float* b2   = (const float*)d_in[8];
    const float* att2 = (const float*)d_in[9];
    const float* W3   = (const float*)d_in[10];
    const float* b3   = (const float*)d_in[11];
    const float* att3 = (const float*)d_in[12];
    const float* Ws3  = (const float*)d_in[13];
    const float* bs3  = (const float*)d_in[14];
    float* outp = (float*)d_out;

    // ---- workspace carve ----
    char* ws = (char*)d_ws;
    auto carve = [&](size_t bytes) { char* p = ws; ws += (bytes + 255) & ~(size_t)255; return p; };
    float*     msg   = (float*)carve((size_t)NN * DH * 4);
    float*     accb  = (float*)carve((size_t)NN * DH * 4);
    float*     hcur  = (float*)carve((size_t)NN * DH * 4);
    _Float16*  Wp1   = (_Float16*)carve((size_t)32 * 16 * 512 * 2);  // K=1024, NCT=16
    _Float16*  Wp2   = (_Float16*)carve((size_t)4  * 8  * 512 * 2);  // K=128,  NCT=8
    _Float16*  Wp3   = (_Float16*)carve((size_t)4  * 2  * 512 * 2);  // K=128,  NCT=2
    float*     score = (float*)carve((size_t)NN * 4);
    float*     amax  = (float*)carve((size_t)NN * 4);
    float*     denom = (float*)carve((size_t)NN * 4);
    float*     eexp  = (float*)carve((size_t)NE * 4);

    const int rowTiles = NN / 16;      // 3125 exactly
    const int gemmBlocks = (rowTiles + 7) / 8;
    const int nodeWaveBlocks = (NN * 32 + 255) / 256;
    const int nodeBlocks = (NN + 255) / 256;
    const int edgeBlocks = (NE + 255) / 256;

    auto run_attn = [&](const float* m, int ldm, int F, const float* att,
                        float* accdst, int ldacc) {
        node_score_kernel<<<nodeWaveBlocks, 256, 0, stream>>>(m, ldm, F, att, score, NN);
        init_attn_kernel<<<nodeBlocks, 256, 0, stream>>>(amax, denom, NN);
        edge_max_kernel<<<edgeBlocks, 256, 0, stream>>>(srcp, dstp, score, amax, NE);
        edge_exp_kernel<<<edgeBlocks, 256, 0, stream>>>(srcp, dstp, score, amax, denom, eexp, NE);
        alpha_kernel<<<edgeBlocks, 256, 0, stream>>>(dstp, denom, eexp, NE);
        int total = NE * F;
        edge_scatter_kernel<<<(total + 255) / 256, 256, 0, stream>>>(
            srcp, dstp, m, ldm, eexp, accdst, ldacc, F, total);
    };

    // ---- pack weights to f16 B-fragment order ----
    pack_b_kernel<<<(32*8*512 + 255)/256, 256, 0, stream>>>(W1,  DIN, DH, Wp1, 16, 0, 8, 32);
    pack_b_kernel<<<(32*8*512 + 255)/256, 256, 0, stream>>>(Ws1, DIN, DH, Wp1, 16, 8, 8, 32);
    pack_b_kernel<<<(4*8*512  + 255)/256, 256, 0, stream>>>(W2,  DH,  DH, Wp2, 8,  0, 8, 4);
    pack_b_kernel<<<(4*1*512  + 255)/256, 256, 0, stream>>>(W3,  DH,  DO, Wp3, 2,  0, 1, 4);
    pack_b_kernel<<<(4*1*512  + 255)/256, 256, 0, stream>>>(Ws3, DH,  DO, Wp3, 2,  1, 1, 4);

    // ================= layer 1 (1024 -> 128) =================
    // fused: msg = X@W1+b1 (cols 0..127), accb = X@Ws1+bs1 (cols 128..255)
    gemm_wmma_kernel<16><<<gemmBlocks, 256, 0, stream>>>(
        x, DIN, 32, rowTiles, Wp1,
        b1, DH, DH, bs1, DH,
        msg, DH, accb, DH);
    run_attn(msg, DH, DH, att1, accb, DH);
    combine_kernel<<<nodeWaveBlocks, 256, 0, stream>>>(accb, hcur, NN);

    // ================= layer 2 (128 -> 128, identity self) =================
    gemm_wmma_kernel<8><<<gemmBlocks, 256, 0, stream>>>(
        hcur, DH, 4, rowTiles, Wp2,
        b2, DH, DH, b2, 0,
        msg, DH, msg, DH);
    hipMemcpyAsync(accb, hcur, (size_t)NN * DH * 4, hipMemcpyDeviceToDevice, stream);
    run_attn(msg, DH, DH, att2, accb, DH);
    combine_kernel<<<nodeWaveBlocks, 256, 0, stream>>>(accb, hcur, NN);

    // ================= layer 3 (128 -> 9) =================
    // tile 0 -> msg (ld 16) = h@W3+b3 ; tile 1 -> accb (ld 16) = h@Ws3+bs3
    gemm_wmma_kernel<2><<<gemmBlocks, 256, 0, stream>>>(
        hcur, DH, 4, rowTiles, Wp3,
        b3, DO, 16, bs3, DO,
        msg, 16, accb, 16);
    run_attn(msg, 16, DO, att3, accb, 16);
    final_kernel<<<nodeBlocks, 256, 0, stream>>>(accb, outp, NN);
}
